// EquivDeepSet_37263136260428
// MI455X (gfx1250) — matrix operand order repaired and checked by reading, not compiled, not run on hardware
//
#include <hip/hip_runtime.h>

typedef __attribute__((ext_vector_type(16))) _Float16 v16h;
typedef __attribute__((ext_vector_type(8)))  _Float16 v8h;
typedef __attribute__((ext_vector_type(8)))  float    v8f;

#define B_SZ   8
#define NCTX   1024
#define G_SZ   16384      // 128 * 128 grid points
#define LOG2E  1.4426950408889634f

// One block: (batch b, 128 grid rows). 8 waves/block, each wave owns a 16-row
// M-tile and contracts Gram(16x1024) x ExpandY(1024x16[3 live cols]) via
// 32 x V_WMMA_F32_16X16X32_F16. Gram is computed on the fly from LDS-staged
// X_c; ExpandY is pre-packed into LDS as f16 in the exact B-operand layout so
// the inner loop is branch-free (2x ds_load_b128 per K-step for B).
__global__ __launch_bounds__(256) void EquivDeepSet_fused_wmma(
    const float* __restrict__ Xc,    // (B, N, 2)
    const float* __restrict__ Yc,    // (B, N, 2)
    const float* __restrict__ grid,  // (G, 2)
    const float* __restrict__ logl,  // (1,)
    float* __restrict__ out)         // (B, 3, G)
{
    __shared__ float    sX[2 * NCTX];       // X_c[b] : 8 KB
    __shared__ _Float16 sBT[16 * NCTX];     // ExpandY^T, f16, col-major: 32 KB

    const int tid  = threadIdx.x;
    const int wave = tid >> 5;
    const int lane = tid & 31;

    const int blk    = blockIdx.x;        // 1024 blocks
    const int b      = blk >> 7;          // batch index
    const int mchunk = blk & 127;         // 128-row chunk within the grid
    const int m0     = mchunk * 128 + wave * 16;  // this wave's 16-row tile

    // ---- stage X_c[b] into LDS (vectorized, coalesced) ----
    {
        const float4* gx4 = (const float4*)(Xc + (size_t)b * (NCTX * 2));
        float4* sx4 = (float4*)sX;
        #pragma unroll
        for (int i = tid; i < (2 * NCTX) / 4; i += 256) {
            sx4[i] = gx4[i];
        }
    }

    // ---- pre-pack ExpandY^T (f16) into LDS: sBT[col*NCTX + k] ----
    // col 0 = 1.0, col 1 = Y[...,0], col 2 = Y[...,1], cols 3..15 = 0.
    {
        const float2* gy2 = (const float2*)(Yc + (size_t)b * (NCTX * 2));
        for (int k = tid; k < NCTX; k += 256) {
            float2 y = gy2[k];
            sBT[0 * NCTX + k] = (_Float16)1.0f;
            sBT[1 * NCTX + k] = (_Float16)y.x;
            sBT[2 * NCTX + k] = (_Float16)y.y;
        }
        // zero-fill dead columns 3..15 with 32-bit stores (offset 6144 B, aligned)
        unsigned int* z32 = (unsigned int*)(sBT + 3 * NCTX);
        #pragma unroll
        for (int i = tid; i < (13 * NCTX) / 2; i += 256) {
            z32[i] = 0u;
        }
    }
    __syncthreads();

    // ---- per-lane constants ----
    const int   half16 = lane >> 4;            // 0: lanes 0-15, 1: lanes 16-31
    const int   row    = m0 + (lane & 15);     // grid row this lane feeds into A
    const float px     = grid[row * 2 + 0];
    const float py     = grid[row * 2 + 1];
    const int   col    = lane & 15;            // B-matrix column for this lane
    const _Float16* myB = sBT + col * NCTX;    // this lane's B column

    float lv = logl[0];
    lv = fminf(fmaxf(lv, -5.0f), 5.0f);
    const float l     = __builtin_amdgcn_exp2f(lv * LOG2E);    // exp(clip(...))
    const float scale = -LOG2E / (2.0f * l * l);               // exp2-domain scale

    v8f acc = {};

    // ---- K loop: 32 context points per WMMA, 32 iterations ----
    for (int K0 = 0; K0 < NCTX; K0 += 32) {
        // A-matrix: Gram entries, K in {0..15} (lanes 0-15) or
        // {16..31} (lanes 16-31), matching the contiguous B fill.
        v16h a;
        const int kaBase = K0 + half16 * 16;
        #pragma unroll
        for (int j = 0; j < 16; ++j) {
            const int ka = kaBase + j;
            float dx = px - sX[2 * ka + 0];
            float dy = py - sX[2 * ka + 1];
            float d2 = __builtin_fmaf(dx, dx, dy * dy);
            a[j] = (_Float16)__builtin_amdgcn_exp2f(d2 * scale);
        }

        // B-matrix: 16 contiguous halves of this lane's column,
        // K in {0..15} (lanes 0-15) or {16..31} (lanes 16-31).
        const int kbBase = K0 + half16 * 16;
        v8h blo = *(const v8h*)(myB + kbBase);
        v8h bhi = *(const v8h*)(myB + kbBase + 8);
        v16h bm = __builtin_shufflevector(blo, bhi,
                    0, 1, 2, 3, 4, 5, 6, 7, 8, 9, 10, 11, 12, 13, 14, 15);

        acc = __builtin_amdgcn_wmma_f32_16x16x32_f16(
            /*neg_a=*/false, a, /*neg_b=*/false, bm,
            /*c_mod=*/(short)0, acc, /*reuse_a=*/false, /*reuse_b=*/false);
    }

    // ---- normalize + store ----
    // C/D layout: VGPR r -> (M = r + 8*half16, N = col). Density lives at N==0,
    // i.e. lane (lane & 16) within the same lane-half. Density is a sum of
    // 1024 positive terms from f16 inputs, so v_rcp_f32 + mul (~1 ulp) is
    // well inside this kernel's precision envelope — avoids the full IEEE
    // v_div_scale/v_div_fmas/v_div_fixup expansion.
    #pragma unroll
    for (int r = 0; r < 8; ++r) {
        float v    = acc[r];
        float dens = __shfl(v, lane & 16, 32);   // FM[m, 0] for this row
        float rden = __builtin_amdgcn_rcpf(dens);
        if (col < 3) {
            int   m = m0 + r + half16 * 8;
            float o = (col == 0) ? v : (v * rden);
            out[((size_t)(b * 3 + col)) * G_SZ + m] = o;
        }
    }
}

extern "C" void kernel_launch(void* const* d_in, const int* in_sizes, int n_in,
                              void* d_out, int out_size, void* d_ws, size_t ws_size,
                              hipStream_t stream) {
    const float* Xc   = (const float*)d_in[0];  // (8, 1024, 2) f32
    const float* Yc   = (const float*)d_in[1];  // (8, 1024, 2) f32
    const float* grid = (const float*)d_in[2];  // (16384, 2)   f32
    const float* logl = (const float*)d_in[3];  // (1,)         f32
    float* out = (float*)d_out;                 // (8, 3, 128, 128) f32

    (void)in_sizes; (void)n_in; (void)out_size; (void)d_ws; (void)ws_size;

    // 1024 blocks = 8 batches * 128 row-chunks; 256 threads = 8 wave32.
    EquivDeepSet_fused_wmma<<<dim3(B_SZ * (G_SZ / 128)), dim3(256), 0, stream>>>(
        Xc, Yc, grid, logl, out);
}
